// ConvexUpsample_31095563223433
// MI455X (gfx1250) — matrix operand order repaired and checked by reading, not compiled, not run on hardware
//
#include <hip/hip_runtime.h>
#include <stdint.h>

typedef __attribute__((ext_vector_type(16))) __bf16 v16bf;
typedef __attribute__((ext_vector_type(8)))  float  v8f;

union Frag {
  uint4 u[2];
  unsigned short h[16];
  v16bf v;
};

#define B_   8
#define C_   128
#define H_   160
#define W_   320
#define HP   162
#define WP   322
#define GROUPS_ 8
#define NELEM_GRP 819200   // (C/G)*H*W = 16*160*320
#define NPIX 51200         // H*W
#define NOUT 48            // conv2 out channels padded 36 -> 48

__device__ __forceinline__ unsigned short f2bf(float f) {
  unsigned u = __float_as_uint(f);
  unsigned r = u + 0x7FFFu + ((u >> 16) & 1u);   // round-to-nearest-even
  return (unsigned short)(r >> 16);
}
__device__ __forceinline__ float bf2f(unsigned short h) {
  return __uint_as_float(((unsigned)h) << 16);
}

// ---------------------------------------------------------------- K0: zero
__global__ void zero_fp_stats(uint4* __restrict__ fp4, size_t n4, float* __restrict__ stats) {
  size_t i = (size_t)blockIdx.x * blockDim.x + threadIdx.x;
  size_t stride = (size_t)gridDim.x * blockDim.x;
  uint4 z; z.x = 0u; z.y = 0u; z.z = 0u; z.w = 0u;
  for (; i < n4; i += stride) fp4[i] = z;
  if (blockIdx.x == 0 && threadIdx.x < 128) stats[threadIdx.x] = 0.f;
}

// --------------------------------------- K1: NCHW f32 -> padded NHWC bf16
__global__ void __launch_bounds__(256) pack_feat(const float* __restrict__ feat,
                                                 unsigned short* __restrict__ fp) {
  __shared__ unsigned short lds[64 * 136];   // 136 row stride: 16B-aligned, conflict-light
  const int b = blockIdx.z, y = blockIdx.y, x0 = blockIdx.x * 64;
  const int t = threadIdx.x;
  const int tx = t & 63, tq = t >> 6;
  for (int c = tq; c < C_; c += 4) {
    float v = feat[(((size_t)b * C_ + c) * H_ + y) * W_ + x0 + tx];
    lds[tx * 136 + c] = f2bf(v);
  }
  __syncthreads();
  const size_t dstbase = ((size_t)(b * HP + y + 1) * WP + (x0 + 1)) * C_;
  #pragma unroll
  for (int i = 0; i < 4; ++i) {
    int ch = i * 256 + t;         // 0..1023 chunks of 8 bf16
    int x  = ch >> 4;
    int c0 = (ch & 15) * 8;
    uint4 v = *(const uint4*)&lds[x * 136 + c0];
    *(uint4*)(fp + dstbase + (size_t)x * C_ + c0) = v;
  }
}

// ------------------------------- K2: pack w1 (OIHW) into B-fragment layout
// frag f = ((tap*4)+kb)*8 + nb ; per-lane 16 bf16 (32B) contiguous
__global__ void pack_w1(const float* __restrict__ w1, unsigned short* __restrict__ w1p) {
  int gid = blockIdx.x * 256 + threadIdx.x;   // 0..9215
  int f = gid >> 5, lane = gid & 31;
  int tap = f >> 5;
  int kb  = (f >> 3) & 3;
  int nb  = f & 7;
  int ky = tap / 3, kx = tap % 3;
  int N = nb * 16 + (lane & 15);
  unsigned short* dst = w1p + ((size_t)f << 9) + (lane << 4);
  #pragma unroll
  for (int e = 0; e < 16; ++e) {
    int K = ((lane >> 4) << 3) + e + ((e >= 8) ? 8 : 0);   // B layout: lanes 16-31 hold K=8..15,24..31
    int I = kb * 32 + K;
    float v = w1[(((size_t)N * C_ + I) * 3 + ky) * 3 + kx];
    dst[e] = f2bf(v);
  }
}

// ------------------------------- K3: pack w2 (36x128) padded to 48 N
__global__ void pack_w2(const float* __restrict__ w2, unsigned short* __restrict__ w2p) {
  int gid = blockIdx.x * 256 + threadIdx.x;
  if (gid >= 12 * 32) return;
  int f = gid >> 5, lane = gid & 31;
  int kb = f / 3, nb = f % 3;
  int N = nb * 16 + (lane & 15);
  unsigned short* dst = w2p + ((size_t)f << 9) + (lane << 4);
  #pragma unroll
  for (int e = 0; e < 16; ++e) {
    int K = ((lane >> 4) << 3) + e + ((e >= 8) ? 8 : 0);
    int c = kb * 32 + K;
    float v = (N < 36) ? w2[(size_t)N * C_ + c] : 0.f;
    dst[e] = f2bf(v);
  }
}

// --------------------- K4: 3x3 conv, implicit-GEMM bf16 WMMA
// Block: 32-pixel M-tile (2 subtiles), 8 waves = 128 out channels.
// A staged once per block via async global->LDS; B fragments reused by both
// M-subtiles, so global loads per WMMA ~= 1.
__global__ void __launch_bounds__(256) conv1_wmma(const unsigned short* __restrict__ fp,
                                                  const uint4* __restrict__ w1p4,
                                                  unsigned short* __restrict__ hbuf,
                                                  float* __restrict__ stats) {
  __shared__ unsigned short stage[3 * 34 * 128];   // 26112 B: 3 rows x 34 px x 128ch
  __shared__ unsigned short dtile[32 * 128];       // 8192 B: D repack

  const int tid = threadIdx.x;
  const int w = tid >> 5;            // wave = output-channel tile = GN group
  const int lane = tid & 31;
  const int m = lane & 15;
  const int hi = (lane >> 4) & 1;

  int id = blockIdx.x;
  const int xt = id % 10;
  const int y  = (id / 10) % H_;
  const int b  = id / (10 * H_);
  const int x0 = xt * 32;

  // ---- stage A rows [y..y+2] x padded-x [x0..x0+33] x 128ch (bf16)
  {
    const unsigned ldsBase = (unsigned)(size_t)(&stage[0]);
    #pragma unroll
    for (int i = 0; i < 7; ++i) {
      int ch = i * 256 + tid;               // 16B chunks, 1632 total (544/row)
      if (ch < 1632) {
        int row = ch / 544;
        int off = ch - row * 544;
        const unsigned short* gsrc =
            fp + ((size_t)(b * HP + y + row) * WP + x0) * C_ + off * 8;
        unsigned ldsoff = ldsBase + (unsigned)(row * 8704 + off * 16);
        unsigned long long ga = (unsigned long long)(size_t)gsrc;
        asm volatile("global_load_async_to_lds_b128 %0, %1, off"
                     :: "v"(ldsoff), "v"(ga) : "memory");
      }
    }
    asm volatile("s_wait_asynccnt 0" ::: "memory");
  }
  __syncthreads();

  v8f acc0 = {0.f, 0.f, 0.f, 0.f, 0.f, 0.f, 0.f, 0.f};
  v8f acc1 = {0.f, 0.f, 0.f, 0.f, 0.f, 0.f, 0.f, 0.f};

  #pragma unroll
  for (int ky = 0; ky < 3; ++ky) {
    #pragma unroll
    for (int kx = 0; kx < 3; ++kx) {
      const int tap = ky * 3 + kx;
      #pragma unroll
      for (int kb = 0; kb < 4; ++kb) {
        const int cb0 = kb * 32 + hi * 8;   // A layout: hi half-wave holds K+8
        Frag A0, A1, Bf;
        const unsigned short* l0 = &stage[(ky * 34 + m + kx) * 128 + cb0];
        const unsigned short* l1 = &stage[(ky * 34 + 16 + m + kx) * 128 + cb0];
        A0.u[0] = *(const uint4*)(l0);
        A0.u[1] = *(const uint4*)(l0 + 16);
        A1.u[0] = *(const uint4*)(l1);
        A1.u[1] = *(const uint4*)(l1 + 16);
        const uint4* bp = w1p4 + ((size_t)((tap * 4 + kb) * 8 + w) << 6) + (lane << 1);
        Bf.u[0] = bp[0];
        Bf.u[1] = bp[1];
        acc0 = __builtin_amdgcn_wmma_f32_16x16x32_bf16(false, A0.v, false, Bf.v,
                                                       (short)0, acc0, false, false);
        acc1 = __builtin_amdgcn_wmma_f32_16x16x32_bf16(false, A1.v, false, Bf.v,
                                                       (short)0, acc1, false, false);
      }
    }
  }

  // GroupNorm partial stats: this wave's 16 channels == group w, 32 pixels
  float s = 0.f, q = 0.f;
  #pragma unroll
  for (int v = 0; v < 8; ++v) {
    float a = acc0[v]; s += a; q += a * a;
    float c = acc1[v]; s += c; q += c * c;
  }
  #pragma unroll
  for (int off = 16; off >= 1; off >>= 1) {
    s += __shfl_xor(s, off);
    q += __shfl_xor(q, off);
  }
  if (lane == 0) {
    atomicAdd(&stats[(b * GROUPS_ + w) * 2 + 0], s);
    atomicAdd(&stats[(b * GROUPS_ + w) * 2 + 1], q);
  }

  // D layout: VGPR v -> M = v + 8*hi, N = lane%16. Repack via LDS, store bf16 NHWC.
  #pragma unroll
  for (int v = 0; v < 8; ++v) {
    int M = v + hi * 8;
    dtile[M * 128 + w * 16 + m]        = f2bf(acc0[v]);
    dtile[(16 + M) * 128 + w * 16 + m] = f2bf(acc1[v]);
  }
  __syncthreads();
  {
    const size_t p0 = (size_t)(b * H_ + y) * W_ + x0;
    #pragma unroll
    for (int i = 0; i < 2; ++i) {
      int ch = i * 256 + tid;           // 512 chunks of 8 bf16
      int M  = ch >> 4;
      int c0 = (ch & 15) * 8;
      uint4 val = *(const uint4*)&dtile[M * 128 + c0];
      *(uint4*)(hbuf + (p0 + M) * C_ + c0) = val;
    }
  }
}

// --------------------- K5: per-(b,c) normalization scale/shift
__global__ void finalize_stats(const float* __restrict__ stats,
                               const float* __restrict__ gamma,
                               const float* __restrict__ beta,
                               float* __restrict__ scaleA,
                               float* __restrict__ shiftA) {
  int i = blockIdx.x * 256 + threadIdx.x;   // 0..1023
  int b = i >> 7, c = i & 127, g = c >> 4;
  float sum = stats[(b * GROUPS_ + g) * 2 + 0];
  float sq  = stats[(b * GROUPS_ + g) * 2 + 1];
  const float inv = 1.f / (float)NELEM_GRP;
  float mean = sum * inv;
  float var  = sq * inv - mean * mean;
  float rs   = rsqrtf(var + 1e-5f);
  float sc = gamma[c] * rs;
  scaleA[i] = sc;
  shiftA[i] = beta[c] - mean * sc;
}

// --------------------- K6: norm + SiLU fused into 1x1 conv (bf16 WMMA)
__global__ void __launch_bounds__(256) conv2_wmma(const unsigned short* __restrict__ hbuf,
                                                  const uint4* __restrict__ w2p4,
                                                  const float* __restrict__ scaleA,
                                                  const float* __restrict__ shiftA,
                                                  float* __restrict__ mbuf) {
  const int tid = threadIdx.x;
  const int w = tid >> 5;
  const int lane = tid & 31;
  const int m = lane & 15;
  const int hi = (lane >> 4) & 1;

  const int tileId = blockIdx.x * 8 + w;     // 0..25599
  const size_t p0 = (size_t)tileId * 16;
  const int b = (int)(p0 / NPIX);
  const float* scb = scaleA + b * C_;
  const float* shb = shiftA + b * C_;
  const unsigned short* hrow = hbuf + (p0 + m) * C_;

  Frag A[4];
  #pragma unroll
  for (int kb = 0; kb < 4; ++kb) {
    const int cb0 = kb * 32 + hi * 8;
    Frag R;
    R.u[0] = *(const uint4*)(hrow + cb0);
    R.u[1] = *(const uint4*)(hrow + cb0 + 16);
    #pragma unroll
    for (int e = 0; e < 16; ++e) {
      int c = cb0 + e + ((e >= 8) ? 8 : 0);  // e<8 -> cb0+e ; e>=8 -> cb0+16+(e-8)
      float x  = bf2f(R.h[e]);
      float xn = x * scb[c] + shb[c];
      float sl = xn / (1.f + __expf(-xn));   // SiLU
      A[kb].h[e] = f2bf(sl);
    }
  }

  #pragma unroll
  for (int nb = 0; nb < 3; ++nb) {
    v8f acc = {0.f, 0.f, 0.f, 0.f, 0.f, 0.f, 0.f, 0.f};
    #pragma unroll
    for (int kb = 0; kb < 4; ++kb) {
      Frag Bf;
      const uint4* bp = w2p4 + ((size_t)(kb * 3 + nb) << 6) + (lane << 1);
      Bf.u[0] = bp[0];
      Bf.u[1] = bp[1];
      acc = __builtin_amdgcn_wmma_f32_16x16x32_bf16(false, A[kb].v, false, Bf.v,
                                                    (short)0, acc, false, false);
    }
    const int N = nb * 16 + m;
    #pragma unroll
    for (int v = 0; v < 8; ++v) {
      int M = v + hi * 8;
      mbuf[(p0 + M) * NOUT + N] = acc[v];
    }
  }
}

// --------------------- K7: bias + softmax(9) x4 + convex disp upsample
__global__ void __launch_bounds__(256) upsample_softmax(const float* __restrict__ mbuf,
                                                        const float* __restrict__ b2,
                                                        const float* __restrict__ disp,
                                                        float* __restrict__ out) {
  int p = blockIdx.x * 256 + threadIdx.x;    // 0..409599 exactly
  int b = p / NPIX;
  int r = p % NPIX;
  int y = r / W_, x = r % W_;

  float v[36];
  const float4* pm = (const float4*)(mbuf + (size_t)p * NOUT);
  #pragma unroll
  for (int j = 0; j < 9; ++j) {
    float4 q4 = pm[j];
    v[4 * j + 0] = q4.x; v[4 * j + 1] = q4.y;
    v[4 * j + 2] = q4.z; v[4 * j + 3] = q4.w;
  }
  #pragma unroll
  for (int j = 0; j < 36; ++j) v[j] += b2[j];

  float d[9];
  const float* db = disp + (size_t)b * NPIX;
  #pragma unroll
  for (int ky = 0; ky < 3; ++ky)
    #pragma unroll
    for (int kx = 0; kx < 3; ++kx) {
      int yy = y + ky - 1, xx = x + kx - 1;
      bool ok = (yy >= 0) && (yy < H_) && (xx >= 0) && (xx < W_);
      d[ky * 3 + kx] = ok ? 2.f * db[yy * W_ + xx] : 0.f;
    }

  float* ob = out + (size_t)b * (4 * NPIX);
  #pragma unroll
  for (int q = 0; q < 4; ++q) {              // q = s1*2+s2 ; m channel = k*4+q
    int s1 = q >> 1, s2 = q & 1;
    float mx = -1e30f;
    #pragma unroll
    for (int k = 0; k < 9; ++k) mx = fmaxf(mx, v[k * 4 + q]);
    float sum = 0.f, o = 0.f;
    #pragma unroll
    for (int k = 0; k < 9; ++k) {
      float e = __expf(v[k * 4 + q] - mx);
      sum += e;
      o += e * d[k];
    }
    ob[(2 * y + s1) * (2 * W_) + 2 * x + s2] = o / sum;
  }
}

// ---------------------------------------------------------------- launcher
extern "C" void kernel_launch(void* const* d_in, const int* in_sizes, int n_in,
                              void* d_out, int out_size, void* d_ws, size_t ws_size,
                              hipStream_t stream) {
  const float* disp  = (const float*)d_in[0];
  const float* feat  = (const float*)d_in[1];
  const float* w1    = (const float*)d_in[2];
  const float* gamma = (const float*)d_in[3];
  const float* beta  = (const float*)d_in[4];
  const float* w2    = (const float*)d_in[5];
  const float* b2    = (const float*)d_in[6];
  float* out = (float*)d_out;
  (void)in_sizes; (void)n_in; (void)out_size; (void)ws_size;

  char* p = (char*)d_ws;
  auto take = [&](size_t bytes) -> char* {
    char* r = p;
    p += (bytes + 255) & ~(size_t)255;
    return r;
  };
  unsigned short* fp   = (unsigned short*)take((size_t)B_ * HP * WP * C_ * 2);  // 106.8 MB
  unsigned short* hbuf = (unsigned short*)take((size_t)B_ * NPIX * C_ * 2);     // 104.9 MB
  float* mbuf          = (float*)take((size_t)B_ * NPIX * NOUT * 4);            //  78.6 MB
  unsigned short* w1p  = (unsigned short*)take(288 * 1024);
  unsigned short* w2p  = (unsigned short*)take(12 * 1024);
  float* stats         = (float*)take(128 * 4);
  float* scaleA        = (float*)take(1024 * 4);
  float* shiftA        = (float*)take(1024 * 4);

  size_t fp_u4 = (size_t)B_ * HP * WP * C_ * 2 / 16;
  zero_fp_stats<<<4096, 256, 0, stream>>>((uint4*)fp, fp_u4, stats);
  pack_feat<<<dim3(5, H_, B_), 256, 0, stream>>>(feat, fp);
  pack_w1<<<36, 256, 0, stream>>>(w1, w1p);
  pack_w2<<<2, 256, 0, stream>>>(w2, w2p);
  conv1_wmma<<<B_ * H_ * (W_ / 32), 256, 0, stream>>>(fp, (const uint4*)w1p, hbuf, stats);
  finalize_stats<<<4, 256, 0, stream>>>(stats, gamma, beta, scaleA, shiftA);
  conv2_wmma<<<(B_ * NPIX / 16) / 8, 256, 0, stream>>>(hbuf, (const uint4*)w2p,
                                                       scaleA, shiftA, mbuf);
  upsample_softmax<<<B_ * NPIX / 256, 256, 0, stream>>>(mbuf, b2, disp, out);
}
